// CausalVariationalCSJDAD_56109452755614
// MI455X (gfx1250) — matrix operand order repaired and checked by reading, not compiled, not run on hardware
//
#include <hip/hip_runtime.h>

// ---------------- CDNA5 (gfx1250) fused CausalVariational SDE kernel ----------------
// wave32, WMMA bf16->f32. One fused kernel, 128 tokens per block, 8 waves,
// each wave owns a 16-row M-tile end-to-end. Weights pre-transposed to [N][K]
// bf16 by a prep kernel into d_ws; staged per-GEMM into a 32KB LDS buffer.

typedef __attribute__((ext_vector_type(16))) __bf16          v16bf;
typedef __attribute__((ext_vector_type(8)))  float           v8f;
typedef __attribute__((ext_vector_type(8)))  unsigned short  ushort8;

#if defined(__HIP_DEVICE_COMPILE__) && !__has_builtin(__builtin_amdgcn_wmma_f32_16x16x32_bf16)
#error "wmma bf16 builtin missing for gfx1250 device pass"
#endif

constexpr int   NTOK    = 64 * 1024;       // B*S
constexpr int   TM      = 128;             // tokens per block
constexpr float DTF     = 0.001f;          // 0.01 / 10
constexpr float SQRT_DT = 0.0316227766016838f;

// workspace (bf16) weight offsets, all matrices stored transposed: [N][K] row-major
constexpr int W_ENC1  = 0;      // [128][128] X@W1
constexpr int W_ENC2  = 16384;  // [128][128] (Wmu | Wlv)
constexpr int W_FILMA = 32768;  // [128][64]  (mu.Wu | sigma.Wu)
constexpr int W_FILMB = 40960;  // [128][64]  (J.Wu  | p.Wu)
constexpr int W_DEC1  = 49152;  // [128][64]
constexpr int W_DEC2  = 57344;  // [128][128]
constexpr int W_DEC3  = 73728;  // [128][128]
constexpr int W_TOTAL = 90112;

// small fp32 params in LDS
constexpr int SP_MU = 0, SP_SG = 704, SP_J = 1408, SP_P = 2112;      // 4 FiLM blocks (704 each)
constexpr int F_WG = 0, F_WB = 256, F_BG = 512, F_BB = 576, F_BU = 640;
constexpr int SP_ENCB1 = 2816, SP_BMU = 2944, SP_BLV = 3008, SP_WE = 3072, SP_BE = 3584;
constexpr int SP_DB1 = 3588, SP_DB2 = 3716, SP_DB3 = 3844, SP_TOTAL = 3972;

constexpr size_t SMEM_BYTES =
    (size_t)(16384 /*sW*/ + 16384 /*sA0*/ + 16384 /*sA1*/ + 8192 /*sU*/) * 2 +
    (size_t)(512 /*sE*/ + SP_TOTAL) * 4;
static_assert(SMEM_BYTES < 320 * 1024, "LDS budget");

__device__ __forceinline__ unsigned short f2bf(float f) {
  union { float f; unsigned int u; } x; x.f = f;
  unsigned int r = x.u + 0x7FFFu + ((x.u >> 16) & 1u);
  return (unsigned short)(r >> 16);
}
__device__ __forceinline__ float bf2f(unsigned short h) {
  union { float f; unsigned int u; } x; x.u = ((unsigned int)h) << 16;
  return x.f;
}

// ---------------- prep: fp32 weights -> transposed bf16 in workspace ----------------
__global__ void prep_weights(const float* __restrict__ enc_W1,
                             const float* __restrict__ Wmu, const float* __restrict__ Wlv,
                             const float* __restrict__ muWu, const float* __restrict__ sgWu,
                             const float* __restrict__ JWu,  const float* __restrict__ pWu,
                             const float* __restrict__ dW1,  const float* __restrict__ dW2,
                             const float* __restrict__ dW3,  unsigned short* __restrict__ ws) {
  int i = blockIdx.x * 256 + threadIdx.x;
  if (i >= W_TOTAL) return;
  float v;
  if (i < W_ENC2) {                    // [n][k] <- enc_W1[k][n], 128x128
    int j = i - W_ENC1; int n = j >> 7, k = j & 127; v = enc_W1[k * 128 + n];
  } else if (i < W_FILMA) {            // (Wmu|Wlv), K=128
    int j = i - W_ENC2; int n = j >> 7, k = j & 127;
    v = (n < 64) ? Wmu[k * 64 + n] : Wlv[k * 64 + (n - 64)];
  } else if (i < W_FILMB) {            // (mu.Wu|sg.Wu), K=64
    int j = i - W_FILMA; int n = j >> 6, k = j & 63;
    v = (n < 64) ? muWu[k * 64 + n] : sgWu[k * 64 + (n - 64)];
  } else if (i < W_DEC1) {             // (J.Wu|p.Wu), K=64
    int j = i - W_FILMB; int n = j >> 6, k = j & 63;
    v = (n < 64) ? JWu[k * 64 + n] : pWu[k * 64 + (n - 64)];
  } else if (i < W_DEC2) {             // dec_W1 [64,128] -> [128][64]
    int j = i - W_DEC1; int n = j >> 6, k = j & 63; v = dW1[k * 128 + n];
  } else if (i < W_DEC3) {
    int j = i - W_DEC2; int n = j >> 7, k = j & 127; v = dW2[k * 128 + n];
  } else {
    int j = i - W_DEC3; int n = j >> 7, k = j & 127; v = dW3[k * 128 + n];
  }
  ws[i] = f2bf(v);
}

// ---------------- fragment loaders (ISA 7.12.2 layouts, wave32) ----------------
// A 16x32 bf16: lane<16 -> row=lane, halves[0..7]=K k0..k0+7, halves[8..15]=K k0+16..k0+23
//               lane>=16 -> row=lane-16, K k0+8..15 and k0+24..31
template <int K>
__device__ __forceinline__ void load_A_frags(const unsigned short* act, int row0, int lane,
                                             v16bf* aF) {
  const unsigned short* rp = act + (row0 + (lane & 15)) * K;
  const int k_hi = (lane >> 4) << 3;  // 0 or 8
#pragma unroll
  for (int ks = 0; ks < K / 32; ++ks) {
    union { ushort8 h[2]; v16bf bf; } t;
    t.h[0] = *(const ushort8*)(rp + ks * 32 + k_hi);
    t.h[1] = *(const ushort8*)(rp + ks * 32 + 16 + k_hi);
    aF[ks] = t.bf;
  }
}
// B 32x16 bf16 from transposed weights wT[N][K]: lane<16 -> col n=lane, K k0..k0+7 and k0+16..k0+23
//                                               lane>=16 -> col n=lane-16, K k0+8..15 and k0+24..31
__device__ __forceinline__ v16bf load_B_frag(const unsigned short* wT, int K, int nt, int ks,
                                             int lane) {
  const unsigned short* p = wT + (nt * 16 + (lane & 15)) * K + ks * 32 + ((lane >> 4) << 3);
  union { ushort8 h[2]; v16bf bf; } t;
  t.h[0] = *(const ushort8*)(p);
  t.h[1] = *(const ushort8*)(p + 16);
  return t.bf;
}

template <int K, int NT>
__device__ __forceinline__ void wave_gemm(const unsigned short* act, const unsigned short* wT,
                                          int row0, int lane, v8f* acc) {
  v16bf aF[K / 32];
  load_A_frags<K>(act, row0, lane, aF);
#pragma unroll
  for (int nt = 0; nt < NT; ++nt) {
    v8f c = {};
#pragma unroll
    for (int ks = 0; ks < K / 32; ++ks) {
      v16bf b = load_B_frag(wT, K, nt, ks, lane);
#if defined(__HIP_DEVICE_COMPILE__)
      c = __builtin_amdgcn_wmma_f32_16x16x32_bf16(false, aF[ks], false, b, (short)0, c,
                                                  false, false);
#else
      (void)b;
#endif
    }
    acc[nt] = c;
  }
}

__device__ __forceinline__ void cpf(float* d, const float* s, int n, int t) {
  for (int i = t; i < n; i += 256) d[i] = s[i];
}
__device__ __forceinline__ void cpw(unsigned short* d, const unsigned short* s, int n_us, int t) {
  const uint4* sv = (const uint4*)s; uint4* dv = (uint4*)d;
  for (int i = t; i < (n_us >> 3); i += 256) dv[i] = sv[i];
}

struct Params {
  const float *X, *eps, *dW;
  const unsigned short* wbf;
  const float *enc_b1, *enc_bmu, *enc_blv, *enc_We, *enc_be;
  const float *mu_bu, *mu_Wg, *mu_bg, *mu_Wb, *mu_bb;
  const float *sg_bu, *sg_Wg, *sg_bg, *sg_Wb, *sg_bb;
  const float *J_bu,  *J_Wg,  *J_bg,  *J_Wb,  *J_bb;
  const float *p_bu,  *p_Wg,  *p_bg,  *p_Wb,  *p_bb;
  const float *dec_b1, *dec_b2, *dec_b3;
  float* out;
};

__launch_bounds__(256)
__global__ void fused_csjd_kernel(Params p) {
  extern __shared__ char smem_raw[];
  unsigned short* sW  = (unsigned short*)smem_raw;  // 16384 (weight stage)
  unsigned short* sA0 = sW + 16384;                 // 16384 (X / d1)
  unsigned short* sA1 = sA0 + 16384;                // 16384 (h / d2)
  unsigned short* sU  = sA1 + 16384;                // 8192  (u bf16)
  float* sE = (float*)(sU + 8192);                  // 512   (E per row)
  float* sP = sE + 512;                             // small params

  const int tid   = threadIdx.x;
  const int wave  = tid >> 5;
  const int lane  = tid & 31;
  const int row0  = wave * 16;
  const int base  = blockIdx.x * TM;
  const int cbase = lane & 15;
  const int rsel  = (lane >> 4) << 3;  // +0 or +8 (C-layout row half)

  // ---- P0: small fp32 params + X tile (bf16) + enc1 weights
  cpf(sP + SP_MU + F_WG, p.mu_Wg, 256, tid); cpf(sP + SP_MU + F_WB, p.mu_Wb, 256, tid);
  cpf(sP + SP_MU + F_BG, p.mu_bg, 64, tid);  cpf(sP + SP_MU + F_BB, p.mu_bb, 64, tid);
  cpf(sP + SP_MU + F_BU, p.mu_bu, 64, tid);
  cpf(sP + SP_SG + F_WG, p.sg_Wg, 256, tid); cpf(sP + SP_SG + F_WB, p.sg_Wb, 256, tid);
  cpf(sP + SP_SG + F_BG, p.sg_bg, 64, tid);  cpf(sP + SP_SG + F_BB, p.sg_bb, 64, tid);
  cpf(sP + SP_SG + F_BU, p.sg_bu, 64, tid);
  cpf(sP + SP_J + F_WG, p.J_Wg, 256, tid);   cpf(sP + SP_J + F_WB, p.J_Wb, 256, tid);
  cpf(sP + SP_J + F_BG, p.J_bg, 64, tid);    cpf(sP + SP_J + F_BB, p.J_bb, 64, tid);
  cpf(sP + SP_J + F_BU, p.J_bu, 64, tid);
  cpf(sP + SP_P + F_WG, p.p_Wg, 256, tid);   cpf(sP + SP_P + F_WB, p.p_Wb, 256, tid);
  cpf(sP + SP_P + F_BG, p.p_bg, 64, tid);    cpf(sP + SP_P + F_BB, p.p_bb, 64, tid);
  cpf(sP + SP_P + F_BU, p.p_bu, 64, tid);
  cpf(sP + SP_ENCB1, p.enc_b1, 128, tid);
  cpf(sP + SP_BMU, p.enc_bmu, 64, tid); cpf(sP + SP_BLV, p.enc_blv, 64, tid);
  cpf(sP + SP_WE, p.enc_We, 512, tid);  cpf(sP + SP_BE, p.enc_be, 4, tid);
  cpf(sP + SP_DB1, p.dec_b1, 128, tid); cpf(sP + SP_DB2, p.dec_b2, 128, tid);
  cpf(sP + SP_DB3, p.dec_b3, 128, tid);

  const float* Xt = p.X + (size_t)base * 128;
  for (int i = tid; i < 4096; i += 256) {
    float4 v = ((const float4*)Xt)[i];
    union { unsigned short s[4]; unsigned long long q; } t;
    t.s[0] = f2bf(v.x); t.s[1] = f2bf(v.y); t.s[2] = f2bf(v.z); t.s[3] = f2bf(v.w);
    ((unsigned long long*)sA0)[i] = t.q;
  }
  cpw(sW, p.wbf + W_ENC1, 16384, tid);
  __syncthreads();

  // ---- P1: h = relu(X @ W1 + b1) -> sA1 (bf16)
  {
    v8f acc[8];
    wave_gemm<128, 8>(sA0, sW, row0, lane, acc);
#pragma unroll
    for (int nt = 0; nt < 8; ++nt) {
      const int c = nt * 16 + cbase;
      const float b = sP[SP_ENCB1 + c];
#pragma unroll
      for (int v = 0; v < 8; ++v) {
        float h = acc[nt][v] + b;
        sA1[(row0 + rsel + v) * 128 + c] = f2bf(h > 0.f ? h : 0.f);
      }
    }
  }
  __syncthreads();

  // ---- P2: stage enc2 weights; logits = h @ We + be (VALU, NE=4); then softmax
  cpw(sW, p.wbf + W_ENC2, 16384, tid);
  for (int pr = tid; pr < 512; pr += 256) {
    const int row = pr >> 2, e = pr & 3;
    float a = sP[SP_BE + e];
    const unsigned short* hr = sA1 + row * 128;
    for (int k = 0; k < 128; ++k) a += bf2f(hr[k]) * sP[SP_WE + k * 4 + e];
    sE[pr] = a;
  }
  __syncthreads();
  if (tid < 128) {
    float4 l = *(float4*)&sE[tid * 4];
    float m = fmaxf(fmaxf(l.x, l.y), fmaxf(l.z, l.w));
    float e0 = __expf(l.x - m), e1 = __expf(l.y - m), e2 = __expf(l.z - m), e3 = __expf(l.w - m);
    float inv = 1.f / (e0 + e1 + e2 + e3);
    sE[tid * 4 + 0] = e0 * inv; sE[tid * 4 + 1] = e1 * inv;
    sE[tid * 4 + 2] = e2 * inv; sE[tid * 4 + 3] = e3 * inv;
  }
  __syncthreads();

  // ---- P3: (mu|lv) = h @ (Wmu|Wlv); u = mu + exp(0.5 lv)*eps   (u stays in regs, bf16 -> sU)
  float u_reg[32];
  {
    v8f acc[8];
    wave_gemm<128, 8>(sA1, sW, row0, lane, acc);
#pragma unroll
    for (int nt = 0; nt < 4; ++nt) {
      const int c = nt * 16 + cbase;
      const float bm = sP[SP_BMU + c], bl = sP[SP_BLV + c];
#pragma unroll
      for (int v = 0; v < 8; ++v) {
        const int r = row0 + rsel + v;
        float mu = acc[nt][v] + bm;
        float lv = acc[nt + 4][v] + bl;
        float u = mu + __expf(0.5f * lv) * p.eps[(size_t)(base + r) * 64 + c];
        u_reg[nt * 8 + v] = u;
        sU[r * 64 + c] = f2bf(u);
      }
    }
  }
  __syncthreads();

  // ---- P4: stage FiLM (mu|sigma) weights
  cpw(sW, p.wbf + W_FILMA, 8192, tid);
  __syncthreads();

  // ---- P5: 10 Euler-Maruyama steps
  for (int it = 0; it < 10; ++it) {
    v8f acc[8];
    wave_gemm<64, 8>(sU, sW, row0, lane, acc);
    const float* dwp = p.dW + (size_t)it * NTOK * 64;
#pragma unroll
    for (int nt = 0; nt < 4; ++nt) {
      const int c = nt * 16 + cbase;
      const float* fm = sP + SP_MU; const float* fs = sP + SP_SG;
      const float mg0 = fm[F_WG + c], mg1 = fm[F_WG + 64 + c], mg2 = fm[F_WG + 128 + c], mg3 = fm[F_WG + 192 + c];
      const float mw0 = fm[F_WB + c], mw1 = fm[F_WB + 64 + c], mw2 = fm[F_WB + 128 + c], mw3 = fm[F_WB + 192 + c];
      const float mbg = fm[F_BG + c], mbb = fm[F_BB + c], mbu = fm[F_BU + c];
      const float sg0 = fs[F_WG + c], sg1 = fs[F_WG + 64 + c], sg2 = fs[F_WG + 128 + c], sg3 = fs[F_WG + 192 + c];
      const float sw0 = fs[F_WB + c], sw1 = fs[F_WB + 64 + c], sw2 = fs[F_WB + 128 + c], sw3 = fs[F_WB + 192 + c];
      const float sbg = fs[F_BG + c], sbb = fs[F_BB + c], sbu = fs[F_BU + c];
#pragma unroll
      for (int v = 0; v < 8; ++v) {
        const int r = row0 + rsel + v;
        float4 E = *(float4*)&sE[r * 4];
        float gm = E.x * mg0 + E.y * mg1 + E.z * mg2 + E.w * mg3 + mbg;
        float bm = E.x * mw0 + E.y * mw1 + E.z * mw2 + E.w * mw3 + mbb;
        float gs = E.x * sg0 + E.y * sg1 + E.z * sg2 + E.w * sg3 + sbg;
        float bs = E.x * sw0 + E.y * sw1 + E.z * sw2 + E.w * sw3 + sbb;
        float drift = (acc[nt][v] + mbu) * gm + bm;
        float vol   = (acc[nt + 4][v] + sbu) * gs + bs;
        float u = u_reg[nt * 8 + v] + drift * DTF +
                  vol * (dwp[(size_t)(base + r) * 64 + c] * SQRT_DT);
        u_reg[nt * 8 + v] = u;
        sU[r * 64 + c] = f2bf(u);
      }
    }
    __syncthreads();
  }

  // ---- P6: jump: U_final = U_CF + GAMMA * FiLM_J(U_CF)*sigmoid(FiLM_p(U_CF))
  cpw(sW, p.wbf + W_FILMB, 8192, tid);
  __syncthreads();
  {
    v8f acc[8];
    wave_gemm<64, 8>(sU, sW, row0, lane, acc);
#pragma unroll
    for (int nt = 0; nt < 4; ++nt) {
      const int c = nt * 16 + cbase;
      const float* fj = sP + SP_J; const float* fp = sP + SP_P;
      const float jg0 = fj[F_WG + c], jg1 = fj[F_WG + 64 + c], jg2 = fj[F_WG + 128 + c], jg3 = fj[F_WG + 192 + c];
      const float jw0 = fj[F_WB + c], jw1 = fj[F_WB + 64 + c], jw2 = fj[F_WB + 128 + c], jw3 = fj[F_WB + 192 + c];
      const float jbg = fj[F_BG + c], jbb = fj[F_BB + c], jbu = fj[F_BU + c];
      const float pg0 = fp[F_WG + c], pg1 = fp[F_WG + 64 + c], pg2 = fp[F_WG + 128 + c], pg3 = fp[F_WG + 192 + c];
      const float pw0 = fp[F_WB + c], pw1 = fp[F_WB + 64 + c], pw2 = fp[F_WB + 128 + c], pw3 = fp[F_WB + 192 + c];
      const float pbg = fp[F_BG + c], pbb = fp[F_BB + c], pbu = fp[F_BU + c];
#pragma unroll
      for (int v = 0; v < 8; ++v) {
        const int r = row0 + rsel + v;
        float4 E = *(float4*)&sE[r * 4];
        float gJ = E.x * jg0 + E.y * jg1 + E.z * jg2 + E.w * jg3 + jbg;
        float bJ = E.x * jw0 + E.y * jw1 + E.z * jw2 + E.w * jw3 + jbb;
        float gp = E.x * pg0 + E.y * pg1 + E.z * pg2 + E.w * pg3 + pbg;
        float bp = E.x * pw0 + E.y * pw1 + E.z * pw2 + E.w * pw3 + pbb;
        float Jv = (acc[nt][v] + jbu) * gJ + bJ;
        float xp = (acc[nt + 4][v] + pbu) * gp + bp;
        float pe = 1.f / (1.f + __expf(-xp));
        float u = u_reg[nt * 8 + v] + 0.5f * (Jv * pe);  // GAMMA = 0.5
        sU[r * 64 + c] = f2bf(u);
      }
    }
  }
  __syncthreads();

  // ---- P7: d1 = relu(U_final @ dec_W1 + b1) -> sA0
  cpw(sW, p.wbf + W_DEC1, 8192, tid);
  __syncthreads();
  {
    v8f acc[8];
    wave_gemm<64, 8>(sU, sW, row0, lane, acc);
#pragma unroll
    for (int nt = 0; nt < 8; ++nt) {
      const int c = nt * 16 + cbase;
      const float b = sP[SP_DB1 + c];
#pragma unroll
      for (int v = 0; v < 8; ++v) {
        float h = acc[nt][v] + b;
        sA0[(row0 + rsel + v) * 128 + c] = f2bf(h > 0.f ? h : 0.f);
      }
    }
  }
  __syncthreads();

  // ---- P8: d2 = relu(d1 @ dec_W2 + b2) -> sA1
  cpw(sW, p.wbf + W_DEC2, 16384, tid);
  __syncthreads();
  {
    v8f acc[8];
    wave_gemm<128, 8>(sA0, sW, row0, lane, acc);
#pragma unroll
    for (int nt = 0; nt < 8; ++nt) {
      const int c = nt * 16 + cbase;
      const float b = sP[SP_DB2 + c];
#pragma unroll
      for (int v = 0; v < 8; ++v) {
        float h = acc[nt][v] + b;
        sA1[(row0 + rsel + v) * 128 + c] = f2bf(h > 0.f ? h : 0.f);
      }
    }
  }
  __syncthreads();

  // ---- P9: out = d2 @ dec_W3 + b3 (fp32 store to global)
  cpw(sW, p.wbf + W_DEC3, 16384, tid);
  __syncthreads();
  {
    v8f acc[8];
    wave_gemm<128, 8>(sA1, sW, row0, lane, acc);
#pragma unroll
    for (int nt = 0; nt < 8; ++nt) {
      const int c = nt * 16 + cbase;
      const float b = sP[SP_DB3 + c];
#pragma unroll
      for (int v = 0; v < 8; ++v) {
        const int r = row0 + rsel + v;
        p.out[(size_t)(base + r) * 128 + c] = acc[nt][v] + b;
      }
    }
  }
}

extern "C" void kernel_launch(void* const* d_in, const int* in_sizes, int n_in,
                              void* d_out, int out_size, void* d_ws, size_t ws_size,
                              hipStream_t stream) {
  (void)in_sizes; (void)n_in; (void)out_size; (void)ws_size;
  auto F = [&](int i) { return (const float*)d_in[i]; };
  unsigned short* wbf = (unsigned short*)d_ws;

  // d_in order: X, eps_U, dW, then params dict flattened:
  // 3 enc_W1, 4 enc_b1, 5 enc_Wmu, 6 enc_bmu, 7 enc_Wlv, 8 enc_blv, 9 enc_We, 10 enc_be,
  // mu:{11 Wu,12 bu,13 Wg,14 bg,15 Wb,16 bb}, sigma:{17..22}, J:{23..28}, p:{29..34},
  // 35 dec_W1, 36 dec_b1, 37 dec_W2, 38 dec_b2, 39 dec_W3, 40 dec_b3
  prep_weights<<<(W_TOTAL + 255) / 256, 256, 0, stream>>>(
      F(3), F(5), F(7), F(11), F(17), F(23), F(29), F(35), F(37), F(39), wbf);

  Params P;
  P.X = F(0); P.eps = F(1); P.dW = F(2); P.wbf = wbf;
  P.enc_b1 = F(4); P.enc_bmu = F(6); P.enc_blv = F(8); P.enc_We = F(9); P.enc_be = F(10);
  P.mu_bu = F(12); P.mu_Wg = F(13); P.mu_bg = F(14); P.mu_Wb = F(15); P.mu_bb = F(16);
  P.sg_bu = F(18); P.sg_Wg = F(19); P.sg_bg = F(20); P.sg_Wb = F(21); P.sg_bb = F(22);
  P.J_bu  = F(24); P.J_Wg  = F(25); P.J_bg  = F(26); P.J_Wb  = F(27); P.J_bb  = F(28);
  P.p_bu  = F(30); P.p_Wg  = F(31); P.p_bg  = F(32); P.p_Wb  = F(33); P.p_bb  = F(34);
  P.dec_b1 = F(36); P.dec_b2 = F(38); P.dec_b3 = F(40);
  P.out = (float*)d_out;

  hipError_t err_ = hipFuncSetAttribute(reinterpret_cast<const void*>(&fused_csjd_kernel),
                                        hipFuncAttributeMaxDynamicSharedMemorySize,
                                        (int)SMEM_BYTES);
  (void)err_;
  fused_csjd_kernel<<<NTOK / TM, 256, SMEM_BYTES, stream>>>(P);
}
